// LogicLayer_68358699483224
// MI455X (gfx1250) — matrix operand order repaired and checked by reading, not compiled, not run on hardware
//
#include <hip/hip_runtime.h>
#include <hip/hip_bf16.h>
#include <stdint.h>

#define IN_DIM          8192
#define TPB             256
#define ROWS_PER_BLOCK  2

typedef __attribute__((ext_vector_type(16))) _Float16 v16h;
typedef __attribute__((ext_vector_type(8)))  float    v8f;
typedef __attribute__((ext_vector_type(4)))  float    f32x4;
typedef __attribute__((ext_vector_type(4)))  int      i32x4;

// GATE_COEF (16x4) padded to 16x16 with zero columns so it can sit directly in
// the B-fragment of a 16x16x32 f16 WMMA (B is 32x16; rows 16..31 are zero).
__constant__ float GPAD[16][16] = {
    {0.f, 0.f, 0.f, 0.f},
    {0.f, 0.f, 0.f, 1.f},
    {0.f, 1.f, 0.f,-1.f},
    {0.f, 1.f, 0.f, 0.f},
    {0.f, 0.f, 1.f,-1.f},
    {0.f, 0.f, 1.f, 0.f},
    {0.f, 1.f, 1.f,-2.f},
    {0.f, 1.f, 1.f,-1.f},
    {1.f,-1.f,-1.f, 1.f},
    {1.f,-1.f,-1.f, 2.f},
    {1.f, 0.f,-1.f, 0.f},
    {1.f, 0.f,-1.f, 1.f},
    {1.f,-1.f, 0.f, 0.f},
    {1.f,-1.f, 0.f, 1.f},
    {1.f, 0.f, 0.f,-1.f},
    {1.f, 0.f, 0.f, 0.f},
};

// ---------------------------------------------------------------------------
// Kernel 1: c[o][0..3] = softmax(weights[o]) @ GATE_COEF via WMMA.
// One wave handles 16 output columns: lanes 0..15 each softmax one row into
// LDS, the wave builds A (16x32 f16, K-padded) / B (32x16 f16, G padded)
// fragments, issues one v_wmma_f32_16x16x32_f16, repacks D through LDS and
// stores one coalesced float4 of coefficients per output column.
// ---------------------------------------------------------------------------
__global__ __launch_bounds__(TPB) void softgate_coef_wmma(
    const float* __restrict__ weights, float* __restrict__ cws)
{
    __shared__ __align__(16) float ptile[TPB / 32][16][16];
    __shared__ __align__(16) float ctile[TPB / 32][16][4];

    const int lane    = threadIdx.x & 31;
    const int wave    = threadIdx.x >> 5;
    const int rowBase = blockIdx.x * (TPB / 32) * 16 + wave * 16;

    if (lane < 16) {
        const f32x4* w4 = (const f32x4*)(weights + (size_t)(rowBase + lane) * 16);
        f32x4 wv[4];
        #pragma unroll
        for (int q = 0; q < 4; ++q) wv[q] = w4[q];
        float v[16];
        float m = -3.402823466e38f;
        #pragma unroll
        for (int q = 0; q < 4; ++q) {
            v[4*q+0] = wv[q].x; v[4*q+1] = wv[q].y;
            v[4*q+2] = wv[q].z; v[4*q+3] = wv[q].w;
        }
        #pragma unroll
        for (int g = 0; g < 16; ++g) m = fmaxf(m, v[g]);
        float s = 0.f;
        #pragma unroll
        for (int g = 0; g < 16; ++g) { v[g] = __expf(v[g] - m); s += v[g]; }
        const float inv = __frcp_rn(s);
        #pragma unroll
        for (int g = 0; g < 16; ++g) ptile[wave][lane][g] = v[g] * inv;
    }
    __syncthreads();

    const int Ml = lane & 15;
    const int hi = lane >> 4;

    // A frag: 16-bit A 16x32 layout. Elements 0..7 hold K = hi*8 + e (the real
    // K=0..15 of p); elements 8..15 (K >= 16) are zero padding.
    const f32x4* pa = (const f32x4*)&ptile[wave][Ml][hi * 8];
    const f32x4 p0 = pa[0];
    const f32x4 p1 = pa[1];
    v16h A;
    A[0] = (_Float16)p0.x; A[1] = (_Float16)p0.y;
    A[2] = (_Float16)p0.z; A[3] = (_Float16)p0.w;
    A[4] = (_Float16)p1.x; A[5] = (_Float16)p1.y;
    A[6] = (_Float16)p1.z; A[7] = (_Float16)p1.w;
    #pragma unroll
    for (int e = 8; e < 16; ++e) A[e] = (_Float16)0.f;

    // B frag: 16-bit B 32x16 layout. Elements 0..7 carry K = hi*8 + e (rows of
    // GPAD), N = lane&15; elements 8..15 (K >= 16) are zero. Branchless so
    // the constant loads stay unconditional (no exec-mask branches).
    const float bmask = (float)(1 - hi);
    v16h Bm;
    #pragma unroll
    for (int e = 0; e < 8; ++e)
        Bm[e] = (_Float16)GPAD[hi * 8 + e][Ml];
    #pragma unroll
    for (int e = 8; e < 16; ++e) Bm[e] = (_Float16)0.f;

    v8f acc = {};
    acc = __builtin_amdgcn_wmma_f32_16x16x32_f16(
        /*neg_a=*/false, A, /*neg_b=*/false, Bm,
        /*c_mod=*/(short)0, acc, /*reuse_a=*/false, /*reuse_b=*/false);

    // D layout: VGPR v -> M = v (+8 for hi lanes), N = lane&15; only N<4 is
    // real. Repack through LDS, then lanes 0..15 store one float4 per column.
    if (Ml < 4) {
        #pragma unroll
        for (int v = 0; v < 8; ++v) ctile[wave][v + hi * 8][Ml] = acc[v];
    }
    __builtin_amdgcn_wave_barrier();   // same-wave LDS ops are in-order
    if (lane < 16) {
        const f32x4 c4 = *(const f32x4*)&ctile[wave][lane][0];
        *(f32x4*)(cws + (size_t)(rowBase + lane) * 4) = c4;
    }
}

// ---------------------------------------------------------------------------
// Kernel 2: gather/eval. Block owns ROWS_PER_BLOCK rows of x in LDS, filled
// with async global->LDS b128 copies (one base address + immediate offsets;
// INST_OFFSET is added to both the LDS and global address). Each thread then
// evaluates 4 consecutive output columns per iteration and streams the result
// with non-temporal 128-bit stores.
// ---------------------------------------------------------------------------
__global__ __launch_bounds__(TPB) void softgate_eval(
    const float* __restrict__ x,
    const int*   __restrict__ idx0,
    const int*   __restrict__ idx1,
    const float* __restrict__ cws,
    float*       __restrict__ out,
    int OUT)
{
    __shared__ __align__(16) float sx[ROWS_PER_BLOCK * IN_DIM];

    const int tid  = threadIdx.x;
    const int row0 = blockIdx.x * ROWS_PER_BLOCK;

    // ---- async copy ROWS_PER_BLOCK rows (64 KB) global -> LDS --------------
    {
        const uint32_t lds0 = (uint32_t)(uintptr_t)(&sx[0]) + (uint32_t)(tid * 16);
        const uint64_t g0   = (uint64_t)(uintptr_t)(x + (size_t)row0 * IN_DIM)
                            + (uint64_t)(tid * 16);
        // ROWS_PER_BLOCK*IN_DIM*4 / 16 B / TPB = 16 b128 copies per thread,
        // 512 B per wave-instruction, offsets 0..61440 fit the 24-bit field.
        asm volatile("global_load_async_to_lds_b128 %0, %1, off offset:%c2" :: "v"(lds0), "v"(g0), "i"( 0 * TPB * 16) : "memory");
        asm volatile("global_load_async_to_lds_b128 %0, %1, off offset:%c2" :: "v"(lds0), "v"(g0), "i"( 1 * TPB * 16) : "memory");
        asm volatile("global_load_async_to_lds_b128 %0, %1, off offset:%c2" :: "v"(lds0), "v"(g0), "i"( 2 * TPB * 16) : "memory");
        asm volatile("global_load_async_to_lds_b128 %0, %1, off offset:%c2" :: "v"(lds0), "v"(g0), "i"( 3 * TPB * 16) : "memory");
        asm volatile("global_load_async_to_lds_b128 %0, %1, off offset:%c2" :: "v"(lds0), "v"(g0), "i"( 4 * TPB * 16) : "memory");
        asm volatile("global_load_async_to_lds_b128 %0, %1, off offset:%c2" :: "v"(lds0), "v"(g0), "i"( 5 * TPB * 16) : "memory");
        asm volatile("global_load_async_to_lds_b128 %0, %1, off offset:%c2" :: "v"(lds0), "v"(g0), "i"( 6 * TPB * 16) : "memory");
        asm volatile("global_load_async_to_lds_b128 %0, %1, off offset:%c2" :: "v"(lds0), "v"(g0), "i"( 7 * TPB * 16) : "memory");
        asm volatile("global_load_async_to_lds_b128 %0, %1, off offset:%c2" :: "v"(lds0), "v"(g0), "i"( 8 * TPB * 16) : "memory");
        asm volatile("global_load_async_to_lds_b128 %0, %1, off offset:%c2" :: "v"(lds0), "v"(g0), "i"( 9 * TPB * 16) : "memory");
        asm volatile("global_load_async_to_lds_b128 %0, %1, off offset:%c2" :: "v"(lds0), "v"(g0), "i"(10 * TPB * 16) : "memory");
        asm volatile("global_load_async_to_lds_b128 %0, %1, off offset:%c2" :: "v"(lds0), "v"(g0), "i"(11 * TPB * 16) : "memory");
        asm volatile("global_load_async_to_lds_b128 %0, %1, off offset:%c2" :: "v"(lds0), "v"(g0), "i"(12 * TPB * 16) : "memory");
        asm volatile("global_load_async_to_lds_b128 %0, %1, off offset:%c2" :: "v"(lds0), "v"(g0), "i"(13 * TPB * 16) : "memory");
        asm volatile("global_load_async_to_lds_b128 %0, %1, off offset:%c2" :: "v"(lds0), "v"(g0), "i"(14 * TPB * 16) : "memory");
        asm volatile("global_load_async_to_lds_b128 %0, %1, off offset:%c2" :: "v"(lds0), "v"(g0), "i"(15 * TPB * 16) : "memory");
        asm volatile("s_wait_asynccnt 0" ::: "memory");
    }
    __syncthreads();

    const i32x4* __restrict__ i0v = (const i32x4*)idx0;
    const i32x4* __restrict__ i1v = (const i32x4*)idx1;
    const f32x4* __restrict__ cv  = (const f32x4*)cws;

    const int ngroups = OUT >> 2;  // float4 column groups (2048)
    #pragma unroll 2
    for (int t = tid; t < ngroups; t += TPB) {
        const i32x4 ia  = i0v[t];
        const i32x4 ib  = i1v[t];
        const f32x4 cc0 = cv[4 * t + 0];
        const f32x4 cc1 = cv[4 * t + 1];
        const f32x4 cc2 = cv[4 * t + 2];
        const f32x4 cc3 = cv[4 * t + 3];

        #pragma unroll
        for (int r = 0; r < ROWS_PER_BLOCK; ++r) {
            const float* sr = &sx[r * IN_DIM];
            const float a0 = sr[ia.x], a1 = sr[ia.y], a2 = sr[ia.z], a3 = sr[ia.w];
            const float b0 = sr[ib.x], b1 = sr[ib.y], b2 = sr[ib.z], b3 = sr[ib.w];

            f32x4 o;
            o.x = fmaf(cc0.w, a0 * b0, fmaf(cc0.z, b0, fmaf(cc0.y, a0, cc0.x)));
            o.y = fmaf(cc1.w, a1 * b1, fmaf(cc1.z, b1, fmaf(cc1.y, a1, cc1.x)));
            o.z = fmaf(cc2.w, a2 * b2, fmaf(cc2.z, b2, fmaf(cc2.y, a2, cc2.x)));
            o.w = fmaf(cc3.w, a3 * b3, fmaf(cc3.z, b3, fmaf(cc3.y, a3, cc3.x)));

            f32x4* op = (f32x4*)(out + (size_t)(row0 + r) * OUT);
            __builtin_nontemporal_store(o, op + t);
        }
    }
}

// ---------------------------------------------------------------------------
extern "C" void kernel_launch(void* const* d_in, const int* in_sizes, int n_in,
                              void* d_out, int out_size, void* d_ws, size_t ws_size,
                              hipStream_t stream)
{
    const float* x       = (const float*)d_in[0];
    const float* weights = (const float*)d_in[1];
    const int*   idx0    = (const int*)d_in[2];
    const int*   idx1    = (const int*)d_in[3];
    float*       out     = (float*)d_out;
    float*       cws     = (float*)d_ws;  // OUT*4 floats = 128 KB scratch

    const int OUT = in_sizes[2];            // 8192
    const int Bn  = in_sizes[0] / IN_DIM;   // 4096

    (void)n_in; (void)out_size; (void)ws_size;

    softgate_coef_wmma<<<OUT / ((TPB / 32) * 16), TPB, 0, stream>>>(weights, cws);
    softgate_eval<<<Bn / ROWS_PER_BLOCK, TPB, 0, stream>>>(x, idx0, idx1, cws, out, OUT);
}
